// OLM_43233140802040
// MI455X (gfx1250) — compile-verified, matrix operand order checked by reading, not run on hardware
//
#include <hip/hip_runtime.h>
#include <hip/hip_bf16.h>

#define HH 14
#define WW 14
#define HW 196          // pixels per image
#define CH 2048         // channels
#define PPAD 208        // 13 tiles * 16 pixels (padded)
#define OUT_STRIDE 32

typedef __attribute__((ext_vector_type(16))) __bf16 v16bf;
typedef __attribute__((ext_vector_type(8)))  float  v8f;

// f32 -> bf16 round-to-nearest-even via bit ops (no reliance on cvt codegen)
static __device__ inline __bf16 f2bf(float x) {
  unsigned u = __builtin_bit_cast(unsigned, x);
  u += 0x7FFFu + ((u >> 16) & 1u);
  unsigned short h = (unsigned short)(u >> 16);
  return __builtin_bit_cast(__bf16, h);
}

// Kernel 1: channel partial sums via WMMA.
// grid = (B, nslice), block = 256 (8 waves). Each wave owns 16-pixel tiles and
// chains v_wmma_f32_16x16x32_bf16 with A = ones so D rows = column(channel) sums.
__global__ void __launch_bounds__(256)
attn_partial_wmma(const float* __restrict__ fms, float* __restrict__ part,
                  int nslice, int cps /* channels per slice */) {
  const int b    = blockIdx.x;
  const int s    = blockIdx.y;
  const int lane = threadIdx.x & 31;
  const int wave = threadIdx.x >> 5;
  const int n     = lane & 15;        // column (pixel within tile)
  const int kbase = lane & 16;        // K-half: lanes 16-31 carry channels +16

  const float* __restrict__ src = fms + (size_t)b * CH * HW;
  const int c0 = s * cps;

  v16bf onesA;
#pragma unroll
  for (int i = 0; i < 16; ++i) onesA[i] = f2bf(1.0f);

  for (int t = wave; t < 13; t += 8) {          // wave-uniform trip count
    const int   p  = t * 16 + n;
    const int   pc = (p < HW) ? p : (HW - 1);   // clamp (keep loads in-bounds)
    const float pm = (p < HW) ? 1.0f : 0.0f;    // zero padding lanes
    v8f acc = {};
    for (int cb = 0; cb < cps; cb += 32) {
      const float* col = src + (size_t)(c0 + cb + kbase) * HW + pc;
      if (cb + 32 < cps) __builtin_prefetch(col + 32 * HW, 0, 0);
      v16bf bm;
#pragma unroll
      for (int k = 0; k < 16; ++k) bm[k] = f2bf(col[k * HW] * pm);
      // D[m][n] = sum_k 1 * B[k][n]  (channel sum per pixel column)
      acc = __builtin_amdgcn_wmma_f32_16x16x32_bf16(
          false, onesA, false, bm, (short)0, acc, false, false);
    }
    if (lane < 16 && p < HW)
      part[((size_t)b * nslice + s) * PPAD + p] = acc[0];   // D[0][n], lane n
  }
}

// Kernel 2: per-image threshold, 8-connected min-label propagation, largest
// component, bbox. One block (256 threads) per image; everything in LDS.
__global__ void __launch_bounds__(256)
finalize_cc(const float* __restrict__ part, int* __restrict__ out, int nslice) {
  const int b = blockIdx.x;
  const int t = threadIdx.x;

  __shared__ float att[HW];
  __shared__ float red[256];
  __shared__ int   lab[HW];
  __shared__ int   msk[HW];
  __shared__ int   counts[HW + 1];
  __shared__ int   bestkey, minr, maxr, minc, maxc;

  if (t == 0) { bestkey = 0; minr = HH; maxr = -1; minc = WW; maxc = -1; }
  if (t <= HW) counts[t] = 0;
  if (t < HW) {
    float ssum = 0.f;
    const float* pp = part + (size_t)b * nslice * PPAD + t;
    for (int sl = 0; sl < nslice; ++sl) ssum += pp[sl * PPAD];
    att[t] = ssum;          // un-normalized mean; scaling cancels in threshold
  }
  __syncthreads();

  red[t] = (t < HW) ? att[t] : -3.0e38f;
  __syncthreads();
  for (int off = 128; off > 0; off >>= 1) {
    if (t < off) red[t] = fmaxf(red[t], red[t + off]);
    __syncthreads();
  }
  const float thr = red[0] * 0.3f;

  if (t < HW) {
    msk[t] = (att[t] > thr) ? 1 : 0;
    lab[t] = msk[t] ? t : HW;        // HW == background sentinel
  }
  __syncthreads();

  // Jacobi min-label propagation, HW iterations = safe upper bound on paths
  for (int it = 0; it < HW; ++it) {
    int nl = HW;
    if (t < HW && msk[t]) {
      const int r = t / WW, c = t % WW;
      nl = lab[t];
#pragma unroll
      for (int dr = -1; dr <= 1; ++dr)
#pragma unroll
        for (int dc = -1; dc <= 1; ++dc) {
          const int rr = r + dr, cc = c + dc;
          if (rr >= 0 && rr < HH && cc >= 0 && cc < WW) {
            const int q = lab[rr * WW + cc];
            nl = (q < nl) ? q : nl;
          }
        }
    }
    __syncthreads();
    if (t < HW && msk[t]) lab[t] = nl;
    __syncthreads();
  }

  if (t < HW) atomicAdd(&counts[lab[t]], 1);
  __syncthreads();
  // argmax(count), tie -> smallest label id (raster-first, as in reference)
  if (t < HW) atomicMax(&bestkey, (counts[t] << 8) | (HW - 1 - t));
  __syncthreads();
  const int best = (HW - 1) - (bestkey & 255);
  const int bcnt = bestkey >> 8;

  if (t < HW && lab[t] == best) {    // background lab==HW can never match
    atomicMin(&minr, t / WW);
    atomicMax(&maxr, t / WW);
    atomicMin(&minc, t % WW);
    atomicMax(&maxc, t % WW);
  }
  __syncthreads();

  if (t == 0) {
    int r0 = minr, r1 = maxr + 1, c0 = minc, c1 = maxc + 1;
    if (bcnt == 0) { r0 = 0; c0 = 0; r1 = HH; c1 = WW; }   // empty-mask fallback
    int* o = out + b * 4;
    o[0] = max(r0 * OUT_STRIDE - 1, 0);
    o[1] = max(c0 * OUT_STRIDE - 1, 0);
    o[2] = r1 * OUT_STRIDE - 1;
    o[3] = c1 * OUT_STRIDE - 1;
  }
}

extern "C" void kernel_launch(void* const* d_in, const int* in_sizes, int n_in,
                              void* d_out, int out_size, void* d_ws, size_t ws_size,
                              hipStream_t stream) {
  const float* fms = (const float*)d_in[0];
  int*   out  = (int*)d_out;
  float* part = (float*)d_ws;

  const int B = in_sizes[0] / (CH * HW);

  // Pick channel-slice count for parallelism, bounded by workspace size.
  int nsl = 16;
  while (nsl > 1 && (size_t)B * nsl * PPAD * sizeof(float) > ws_size) nsl >>= 1;
  const int cps = CH / nsl;

  dim3 g1((unsigned)B, (unsigned)nsl);
  attn_partial_wmma<<<g1, 256, 0, stream>>>(fms, part, nsl, cps);
  finalize_cc<<<B, 256, 0, stream>>>(part, out, nsl);
}
